// WindowMSA_14018773254452
// MI455X (gfx1250) — matrix operand order repaired and checked by reading, not compiled, not run on hardware
//
#include <hip/hip_runtime.h>

// ---------------------------------------------------------------------------
// MI455X (gfx1250) window-MSA. All four matmuls run on v_wmma_f32_16x16x32_f16
// (f16 inputs, fp32 accum). Compute-bound problem (~200 GFLOP / ~1 GB), so the
// matrix pipe is the target; HBM at 23.3 TB/s easily feeds the tiles.
// ---------------------------------------------------------------------------

typedef _Float16 h16;
typedef __attribute__((ext_vector_type(16))) _Float16 v16h;
typedef __attribute__((ext_vector_type(8)))  _Float16 h8v;
typedef __attribute__((ext_vector_type(8)))  float    v8f;

#define DEV __device__ __forceinline__

// Problem constants
#define NB   192
#define NHW  196          // N = 14*14
#define NC   768
#define NHEAD 12
#define DH   64
#define BH   (NB*NHEAD)   // 2304
#define M_ROWS (NB*NHW)   // 37632
#define NPAD 208          // key rows padded to 13*16
#define PPAD 224          // P columns padded to 7*32

DEV v8f wmma_f16(v16h a, v16h b, v8f c) {
  return __builtin_amdgcn_wmma_f32_16x16x32_f16(false, a, false, b, (short)0, c,
                                                false, false);
}

// A-fragment (16x32, MxK), row-major source, K-contiguous rows.
// lane: r=lane&15 (row), hi=lane>>4. Element e: K = hi*16 + e.
DEV v16h load_a_frag(const h16* base, int ld, int kb, int lane) {
  int r = lane & 15, hi = lane >> 4;
  const h16* p = base + (size_t)r * ld + kb;
  h8v lo = *(const h8v*)(p + 16 * hi);
  h8v hh = *(const h8v*)(p + 16 * hi + 8);
  return __builtin_shufflevector(lo, hh, 0,1,2,3,4,5,6,7,8,9,10,11,12,13,14,15);
}

// B-fragment (32x16, KxN). Column n of B stored K-contiguous (row n of source).
// lane: n=lane&15, hi=lane>>4. Element e: K = hi*16 + e -> one 32-byte load.
DEV v16h load_b_frag(const h16* base, int ld, int kb, int lane) {
  int n = lane & 15, hi = lane >> 4;
  return *(const v16h*)(base + (size_t)n * ld + kb + 16 * hi);
}

// ---------------------------------------------------------------------------
__global__ __launch_bounds__(256) void f32_to_f16_kernel(const float* __restrict__ in,
                                                         h16* __restrict__ out,
                                                         size_t n) {
  size_t i = (size_t)blockIdx.x * blockDim.x + threadIdx.x;
  size_t stride = (size_t)gridDim.x * blockDim.x;
  for (; i < n; i += stride) out[i] = (h16)in[i];
}

// ---------------------------------------------------------------------------
// qkv GEMM: (37632 x 768) @ (2304 x 768)^T. One wave computes 32(M) x 64(N).
// Epilogue: +bias, scatter to qh (x0.125, padded), kh (padded), vT (transposed).
__global__ __launch_bounds__(256) void qkv_gemm_kernel(
    const h16* __restrict__ xh, const h16* __restrict__ wh,
    const float* __restrict__ qkv_b,
    h16* __restrict__ qh, h16* __restrict__ kh, h16* __restrict__ vT) {
  int wave = blockIdx.x * 8 + (threadIdx.x >> 5);
  int lane = threadIdx.x & 31;
  int nt = wave / 1176;            // 0..35
  int mt = wave - nt * 1176;       // 0..1175
  int m0 = mt * 32, n0 = nt * 64;

  v8f acc[2][4] = {};
  for (int kb = 0; kb < NC; kb += 32) {
    v16h a0 = load_a_frag(xh + (size_t)m0 * NC, NC, kb, lane);
    v16h a1 = load_a_frag(xh + (size_t)(m0 + 16) * NC, NC, kb, lane);
#pragma unroll
    for (int j = 0; j < 4; ++j) {
      v16h bf = load_b_frag(wh + (size_t)(n0 + 16 * j) * NC, NC, kb, lane);
      acc[0][j] = wmma_f16(a0, bf, acc[0][j]);
      acc[1][j] = wmma_f16(a1, bf, acc[1][j]);
    }
  }

  int cn = lane & 15, hi = lane >> 4;
#pragma unroll
  for (int i = 0; i < 2; ++i)
#pragma unroll
    for (int j = 0; j < 4; ++j) {
      int col3 = n0 + 16 * j + cn;           // 0..2303
      int which = col3 / NC;                 // 0=q,1=k,2=v
      int rem = col3 - which * NC;
      int head = rem >> 6, c = rem & 63;
      float bias = qkv_b[col3];
#pragma unroll
      for (int r = 0; r < 8; ++r) {
        int m = m0 + 16 * i + r + 8 * hi;
        int batch = m / NHW, n = m - batch * NHW;
        int bh = batch * NHEAD + head;
        float val = acc[i][j][r] + bias;
        if (which == 0)      qh[((size_t)bh * NPAD + n) * DH + c] = (h16)(val * 0.125f);
        else if (which == 1) kh[((size_t)bh * NPAD + n) * DH + c] = (h16)val;
        else                 vT[((size_t)bh * DH + c) * PPAD + n] = (h16)val;
      }
    }
}

// ---------------------------------------------------------------------------
// Decomposed rel-pos terms: relH[b,n,j]=qh[b,n,:].Rh[n/14 - j + 13,:], same for W.
__global__ __launch_bounds__(256) void relpos_kernel(
    const h16* __restrict__ qh, const float* __restrict__ rph,
    const float* __restrict__ rpw,
    float* __restrict__ relHbuf, float* __restrict__ relWbuf) {
  size_t tid = (size_t)blockIdx.x * 256 + threadIdx.x;   // BH*NHW*28 total
  int j28 = (int)(tid % 28);
  size_t bn = tid / 28;
  int n = (int)(bn % NHW);
  int hq = n / 14, wq = n - hq * 14;
  bool isH = j28 < 14;
  int j = isH ? j28 : j28 - 14;
  const float* R = isH ? (rph + (size_t)(hq - j + 13) * DH)
                       : (rpw + (size_t)(wq - j + 13) * DH);
  size_t b = bn / NHW;
  const h16* qp = qh + ((size_t)b * NPAD + n) * DH;
  float s = 0.f;
#pragma unroll 8
  for (int c = 0; c < DH; ++c) s += (float)qp[c] * R[c];
  (isH ? relHbuf : relWbuf)[bn * 14 + j] = s;
}

// ---------------------------------------------------------------------------
// Attention: one wave per (b, 16-row q tile). Scores (16x208) in 13 v8f accums,
// +0.125 scale + rel bias, softmax, P->LDS f16, P@V (16x224 @ 224x64).
__global__ __launch_bounds__(32) void attn_kernel(
    const h16* __restrict__ qh, const h16* __restrict__ kh,
    const h16* __restrict__ vT,
    const float* __restrict__ relHbuf, const float* __restrict__ relWbuf,
    h16* __restrict__ aoh) {
  const int mt = blockIdx.x;   // 0..12
  const int b  = blockIdx.y;   // 0..2303
  const int lane = threadIdx.x;
  const int cn = lane & 15, hi = lane >> 4;

  __shared__ h16 Plds[16 * PPAD];
  __shared__ float relS[2][16][14];

  {   // stage rel bias for the 16 q rows (lanes 0-15: H, 16-31: W)
    const float* rb = (hi == 0) ? relHbuf : relWbuf;
    int qr = mt * 16 + cn;
    const float* src = rb + ((size_t)b * NHW + (qr < NHW ? qr : 0)) * 14;
#pragma unroll
    for (int j = 0; j < 14; ++j)
      relS[hi][cn][j] = (qr < NHW) ? src[j] : 0.f;
  }
  __syncthreads();

  const h16* qbase = qh + ((size_t)b * NPAD + mt * 16) * DH;
  v16h a0 = load_a_frag(qbase, DH, 0, lane);
  v16h a1 = load_a_frag(qbase, DH, 32, lane);

  v8f s[13];
#pragma unroll
  for (int t = 0; t < 13; ++t) {
    const h16* kb0 = kh + ((size_t)b * NPAD + t * 16) * DH;
    v8f acc = {};
    acc = wmma_f16(a0, load_b_frag(kb0, DH, 0, lane), acc);
    acc = wmma_f16(a1, load_b_frag(kb0, DH, 32, lane), acc);
    s[t] = acc;
  }

  // second scale (reference double-scales q) + rel bias + pad mask
#pragma unroll
  for (int t = 0; t < 13; ++t) {
    int kn = t * 16 + cn;
    if (kn < NHW) {
      int hk = kn / 14, wk = kn - hk * 14;
#pragma unroll
      for (int r = 0; r < 8; ++r) {
        int m = r + 8 * hi;
        s[t][r] = s[t][r] * 0.125f + relS[0][m][hk] + relS[1][m][wk];
      }
    } else {
#pragma unroll
      for (int r = 0; r < 8; ++r) s[t][r] = -1e30f;
    }
  }

  // row softmax: row m = vgpr + 8*(lane>=16); 16 cols per tile live in lanes
  float mx[8], sm[8];
#pragma unroll
  for (int r = 0; r < 8; ++r) {
    float m = s[0][r];
#pragma unroll
    for (int t = 1; t < 13; ++t) m = fmaxf(m, s[t][r]);
#pragma unroll
    for (int off = 1; off < 16; off <<= 1) m = fmaxf(m, __shfl_xor(m, off, 32));
    mx[r] = m;
    sm[r] = 0.f;
  }
#pragma unroll
  for (int t = 0; t < 13; ++t)
#pragma unroll
    for (int r = 0; r < 8; ++r) {
      float e = __expf(s[t][r] - mx[r]);
      s[t][r] = e;
      sm[r] += e;
    }
#pragma unroll
  for (int r = 0; r < 8; ++r) {
    float v = sm[r];
#pragma unroll
    for (int off = 1; off < 16; off <<= 1) v += __shfl_xor(v, off, 32);
    sm[r] = 1.f / v;
  }

  // P -> LDS (f16), zero the 208..223 pad columns
#pragma unroll
  for (int t = 0; t < 13; ++t)
#pragma unroll
    for (int r = 0; r < 8; ++r)
      Plds[(r + 8 * hi) * PPAD + t * 16 + cn] = (h16)(s[t][r] * sm[r]);
#pragma unroll
  for (int r = 0; r < 8; ++r)
    Plds[(r + 8 * hi) * PPAD + 208 + cn] = (h16)0.f;
  __syncthreads();

  // out = P (16x224) @ V (224x64): B columns = rows of vT, K-contiguous.
  v8f o[4] = {};
#pragma unroll
  for (int ks = 0; ks < 7; ++ks) {
    v16h pa = load_a_frag(Plds, PPAD, ks * 32, lane);
#pragma unroll
    for (int j = 0; j < 4; ++j) {
      const h16* vb0 = vT + ((size_t)b * DH + j * 16) * PPAD;
      o[j] = wmma_f16(pa, load_b_frag(vb0, PPAD, ks * 32, lane), o[j]);
    }
  }

  int batch = b / NHEAD, head = b - batch * NHEAD;
#pragma unroll
  for (int j = 0; j < 4; ++j)
#pragma unroll
    for (int r = 0; r < 8; ++r) {
      int qrow = mt * 16 + r + 8 * hi;
      if (qrow < NHW)
        aoh[((size_t)(batch * NHW + qrow)) * NC + head * DH + j * 16 + cn] =
            (h16)o[j][r];
    }
}

// ---------------------------------------------------------------------------
// proj GEMM: (37632 x 768) @ (768 x 768)^T + bias -> fp32 output.
__global__ __launch_bounds__(256) void proj_gemm_kernel(
    const h16* __restrict__ aoh, const h16* __restrict__ pwh,
    const float* __restrict__ pb, float* __restrict__ out) {
  int wave = blockIdx.x * 8 + (threadIdx.x >> 5);
  int lane = threadIdx.x & 31;
  int nt = wave / 1176;            // 0..11
  int mt = wave - nt * 1176;
  int m0 = mt * 32, n0 = nt * 64;

  v8f acc[2][4] = {};
  for (int kb = 0; kb < NC; kb += 32) {
    v16h a0 = load_a_frag(aoh + (size_t)m0 * NC, NC, kb, lane);
    v16h a1 = load_a_frag(aoh + (size_t)(m0 + 16) * NC, NC, kb, lane);
#pragma unroll
    for (int j = 0; j < 4; ++j) {
      v16h bf = load_b_frag(pwh + (size_t)(n0 + 16 * j) * NC, NC, kb, lane);
      acc[0][j] = wmma_f16(a0, bf, acc[0][j]);
      acc[1][j] = wmma_f16(a1, bf, acc[1][j]);
    }
  }

  int cn = lane & 15, hi = lane >> 4;
#pragma unroll
  for (int i = 0; i < 2; ++i)
#pragma unroll
    for (int j = 0; j < 4; ++j) {
      int col = n0 + 16 * j + cn;
      float bias = pb[col];
#pragma unroll
      for (int r = 0; r < 8; ++r) {
        int m = m0 + 16 * i + r + 8 * hi;
        out[(size_t)m * NC + col] = acc[i][j][r] + bias;
      }
    }
}

// ---------------------------------------------------------------------------
extern "C" void kernel_launch(void* const* d_in, const int* in_sizes, int n_in,
                              void* d_out, int out_size, void* d_ws, size_t ws_size,
                              hipStream_t stream) {
  const float* x      = (const float*)d_in[0];
  const float* qkv_w  = (const float*)d_in[1];
  const float* qkv_b  = (const float*)d_in[2];
  const float* proj_w = (const float*)d_in[3];
  const float* proj_b = (const float*)d_in[4];
  const float* rph    = (const float*)d_in[5];
  const float* rpw    = (const float*)d_in[6];
  float* out = (float*)d_out;

  const size_t XH_B   = (size_t)M_ROWS * NC * 2;        // 57,802,752
  const size_t WH_B   = (size_t)3 * NC * NC * 2;        //  3,538,944
  const size_t PWH_B  = (size_t)NC * NC * 2;            //  1,179,648
  const size_t QH_B   = (size_t)BH * NPAD * DH * 2;     // 61,341,696
  const size_t VT_B   = (size_t)BH * DH * PPAD * 2;     // 66,060,288
  const size_t REL_B  = (size_t)BH * NHW * 14 * 4;      // 25,288,704

  char* ws = (char*)d_ws;
  size_t off = 0;
  auto take = [&](size_t bytes) {
    char* p = ws + off;
    off += (bytes + 255) & ~(size_t)255;
    return p;
  };
  h16*   xh      = (h16*)take(XH_B);     // reused as attention output (aoh)
  h16*   wh      = (h16*)take(WH_B);
  h16*   pwh     = (h16*)take(PWH_B);
  h16*   qh      = (h16*)take(QH_B);
  h16*   kh      = (h16*)take(QH_B);
  h16*   vT      = (h16*)take(VT_B);
  float* relHbuf = (float*)take(REL_B);
  float* relWbuf = (float*)take(REL_B);
  h16*   aoh     = xh;   // x is dead after the qkv GEMM; alias its buffer

  (void)in_sizes; (void)n_in; (void)out_size; (void)ws_size;

  // Zero padded q/k/v storage so pad rows are benign (graph-capturable).
  hipMemsetAsync(qh, 0, QH_B, stream);
  hipMemsetAsync(kh, 0, QH_B, stream);
  hipMemsetAsync(vT, 0, VT_B, stream);

  // fp32 -> f16 staging
  f32_to_f16_kernel<<<2048, 256, 0, stream>>>(x, xh, (size_t)M_ROWS * NC);
  f32_to_f16_kernel<<<512,  256, 0, stream>>>(qkv_w, wh, (size_t)3 * NC * NC);
  f32_to_f16_kernel<<<256,  256, 0, stream>>>(proj_w, pwh, (size_t)NC * NC);

  // qkv projection (WMMA), scatter into q/k/vT
  qkv_gemm_kernel<<<5292, 256, 0, stream>>>(xh, wh, qkv_b, qh, kh, vT);

  // decomposed relative-position dot products
  relpos_kernel<<<49392, 256, 0, stream>>>(qh, rph, rpw, relHbuf, relWbuf);

  // fused attention (scores WMMA + bias + softmax + P@V WMMA)
  attn_kernel<<<dim3(13, BH), 32, 0, stream>>>(qh, kh, vT, relHbuf, relWbuf, aoh);

  // output projection (WMMA) + bias -> fp32 result
  proj_gemm_kernel<<<1764, 256, 0, stream>>>(aoh, pwh, proj_b, out);
}
